// MAB_28656021799628
// MI455X (gfx1250) — matrix operand order repaired and checked by reading, not compiled, not run on hardware
//
#include <hip/hip_runtime.h>
#include <hip/hip_bf16.h>

typedef __attribute__((ext_vector_type(16))) _Float16 v16h;
typedef __attribute__((ext_vector_type(8)))  _Float16 v8h;
typedef __attribute__((ext_vector_type(8)))  float    v8f;
typedef __attribute__((ext_vector_type(4)))  int      v4i;

// Problem constants (fixed by the reference).
#define BB   4
#define NQ   2048
#define NK   2048
#define DD   256
#define FF   1024
#define HH   8
#define DH   32
#define NTOK (BB * NQ)

// CDNA5 async global->LDS copy support (guarded; falls back to plain LDS store)
#ifndef __has_builtin
#define __has_builtin(x) 0
#endif
#if __has_builtin(__builtin_amdgcn_global_load_async_to_lds_b128) && \
    __has_builtin(__builtin_amdgcn_s_wait_asynccnt)
#define HAVE_ASYNC_LDS 1
typedef __attribute__((address_space(1))) v4i* gv4i_p;
typedef __attribute__((address_space(3))) v4i* lv4i_p;
#else
#define HAVE_ASYNC_LDS 0
#endif

// ---------------------------------------------------------------------------
// WMMA helpers
// ---------------------------------------------------------------------------
__device__ __forceinline__ v8f wmma_f16(v16h a, v16h b, v8f c) {
    return __builtin_amdgcn_wmma_f32_16x16x32_f16(false, a, false, b,
                                                  (short)0, c, false, false);
}

// A-fragment: 16x32 f16, row-major source with leading dim `ld`.
// Lanes 0-15 (row M=lane) hold K={0..7,16..23}; lanes 16-31 hold K={8..15,24..31}.
__device__ __forceinline__ v16h fragA_ld(const _Float16* base, int ld,
                                         int mbase, int kbase) {
    int lane = threadIdx.x & 31;
    int m    = lane & 15;
    int k0   = kbase + ((lane & 16) ? 8 : 0);
    const _Float16* p = base + (size_t)(mbase + m) * ld + k0;
    v8h lo = *(const v8h*)p;
    v8h hi = *(const v8h*)(p + 16);
    v16h f;
#pragma unroll
    for (int i = 0; i < 8; ++i) { f[i] = lo[i]; f[i + 8] = hi[i]; }
    return f;
}

// B-fragment: 32x16 f16 from B^T (N x K row-major, leading dim `ld`).
// Lanes 0-15 (col N=lane) hold K={0..7,16..23}; lanes 16-31 hold K={8..15,24..31}.
__device__ __forceinline__ v16h fragB_ld(const _Float16* baseT, int ld,
                                         int nbase, int kbase) {
    int lane = threadIdx.x & 31;
    int n    = lane & 15;
    int k0   = kbase + ((lane & 16) ? 8 : 0);
    const _Float16* p = baseT + (size_t)(nbase + n) * ld + k0;
    v8h lo = *(const v8h*)p;
    v8h hi = *(const v8h*)(p + 16);
    v16h f;
#pragma unroll
    for (int i = 0; i < 8; ++i) { f[i] = lo[i]; f[i + 8] = hi[i]; }
    return f;
}

// D-matrix coordinates for this lane: col n, row offset moff (+r per VGPR).
__device__ __forceinline__ void dcoords(int& n, int& moff) {
    int lane = threadIdx.x & 31;
    n    = lane & 15;
    moff = (lane & 16) ? 8 : 0;
}

// ---------------------------------------------------------------------------
// Blocked GEMM core: 128-thread block computes a 64x64 tile of
// A(MxK,row-major) * B^T(NxK,row-major). Wave w owns rows [w*16, w*16+16);
// the 64xK B strip is staged through double-buffered LDS (async DMA when
// available) and shared by all 4 waves. Each K-chunk: 1 A-frag load,
// 4 LDS B-frags, 4 WMMAs.
// ---------------------------------------------------------------------------
__device__ __forceinline__ void stage_B(const _Float16* src, int ldk,
                                        _Float16* dstLDS) {
    // copy 64 rows x 32 halves (4KB) with 128 threads, 16B segments x2
    int t = threadIdx.x;
#pragma unroll
    for (int j = 0; j < 2; ++j) {
        int idx = t + 128 * j;
        int row = idx >> 2, seg = idx & 3;
        const _Float16* s = src + (size_t)row * ldk + seg * 8;
        _Float16*       d = dstLDS + row * 32 + seg * 8;
#if HAVE_ASYNC_LDS
        __builtin_amdgcn_global_load_async_to_lds_b128((gv4i_p)s, (lv4i_p)d, 0, 0);
#else
        *(v8h*)d = *(const v8h*)s;
#endif
    }
}

__device__ __forceinline__ void sync_stage() {
#if HAVE_ASYNC_LDS
    __builtin_amdgcn_s_wait_asynccnt(0);
#endif
    __syncthreads();
}

__device__ __forceinline__ void gemm_block(const _Float16* A,
                                           const _Float16* Bt, int K,
                                           int mtile, int ntile0, v8f acc[4]) {
    __shared__ __align__(16) _Float16 Bsh[2 * 64 * 32];
    acc[0] = v8f{}; acc[1] = v8f{}; acc[2] = v8f{}; acc[3] = v8f{};

    const _Float16* Bbase = Bt + (size_t)ntile0 * K;
    stage_B(Bbase, K, Bsh);            // chunk 0 -> buffer 0
    sync_stage();

    for (int k = 0; k < K; k += 32) {
        int cur = (k >> 5) & 1;
        if (k + 32 < K) {
            stage_B(Bbase + (k + 32), K, Bsh + (cur ^ 1) * (64 * 32));
            __builtin_prefetch(A + (size_t)mtile * K + k + 32, 0, 1);
        }
        v16h a = fragA_ld(A, K, mtile, k);
        _Float16* Bcur = Bsh + cur * (64 * 32);
#pragma unroll
        for (int j = 0; j < 4; ++j) {
            v16h b = fragB_ld(Bcur, 32, j * 16, 0);
            acc[j] = wmma_f16(a, b, acc[j]);
        }
        sync_stage();
    }
}

// ---------------------------------------------------------------------------
// Conversion kernels
// ---------------------------------------------------------------------------
__global__ void k_cvt_f16(const float* __restrict__ s, _Float16* __restrict__ d,
                          size_t n) {
    size_t i = (size_t)blockIdx.x * blockDim.x + threadIdx.x;
    if (i < n) d[i] = (_Float16)s[i];
}

// dst[c*R + r] = (f16) src[r*C + c]   (store W^T for contiguous B-fragments)
__global__ void k_tpose_f16(const float* __restrict__ s, _Float16* __restrict__ d,
                            int R, int C) {
    int c = blockIdx.x * 16 + threadIdx.x;
    int r = blockIdx.y * 16 + threadIdx.y;
    if (r < R && c < C) d[(size_t)c * R + r] = (_Float16)s[(size_t)r * C + c];
}

// ---------------------------------------------------------------------------
// GEMM epilogue variants (block = 128 threads = 4 waves; 64x64 output tile)
// ---------------------------------------------------------------------------
__global__ __launch_bounds__(128)
void k_gemm_bias_f16(const _Float16* __restrict__ A,
                     const _Float16* __restrict__ Bt,
                     const float* __restrict__ bias,
                     _Float16* __restrict__ C, int M, int N, int K) {
    int mtile  = blockIdx.y * 64 + (threadIdx.x >> 5) * 16;
    int ntile0 = blockIdx.x * 64;
    v8f acc[4];
    gemm_block(A, Bt, K, mtile, ntile0, acc);
    int n, moff; dcoords(n, moff);
#pragma unroll
    for (int j = 0; j < 4; ++j) {
        float bv = bias[ntile0 + j * 16 + n];
#pragma unroll
        for (int r = 0; r < 8; ++r)
            C[(size_t)(mtile + moff + r) * N + ntile0 + j * 16 + n] =
                (_Float16)(acc[j][r] + bv);
    }
}

// V projection stored transposed: Vt[((b*H + h)*DH + d)*NK + key]
__global__ __launch_bounds__(128)
void k_gemm_vT(const _Float16* __restrict__ A, const _Float16* __restrict__ Bt,
               const float* __restrict__ bias, _Float16* __restrict__ Vt,
               int M, int N, int K) {
    int mtile  = blockIdx.y * 64 + (threadIdx.x >> 5) * 16;
    int ntile0 = blockIdx.x * 64;
    v8f acc[4];
    gemm_block(A, Bt, K, mtile, ntile0, acc);
    int n, moff; dcoords(n, moff);
#pragma unroll
    for (int j = 0; j < 4; ++j) {
        int nn = ntile0 + j * 16 + n;
        int h = nn >> 5, d = nn & 31;
        float bv = bias[nn];
#pragma unroll
        for (int r = 0; r < 8; ++r) {
            int m   = mtile + moff + r;
            int b   = m >> 11;            // / NK
            int key = m & 2047;           // % NK
            Vt[(((size_t)b * HH + h) * DH + d) * NK + key] =
                (_Float16)(acc[j][r] + bv);
        }
    }
}

// X2 = relu(acc + bias) + R32
__global__ __launch_bounds__(128)
void k_gemm_relu_res(const _Float16* __restrict__ A,
                     const _Float16* __restrict__ Bt,
                     const float* __restrict__ bias,
                     const float* __restrict__ R32,
                     float* __restrict__ C, int M, int N, int K) {
    int mtile  = blockIdx.y * 64 + (threadIdx.x >> 5) * 16;
    int ntile0 = blockIdx.x * 64;
    v8f acc[4];
    gemm_block(A, Bt, K, mtile, ntile0, acc);
    int n, moff; dcoords(n, moff);
#pragma unroll
    for (int j = 0; j < 4; ++j) {
        float bv = bias[ntile0 + j * 16 + n];
#pragma unroll
        for (int r = 0; r < 8; ++r) {
            size_t idx = (size_t)(mtile + moff + r) * N + ntile0 + j * 16 + n;
            C[idx] = fmaxf(acc[j][r] + bv, 0.f) + R32[idx];
        }
    }
}

// H = (f16) relu(acc + bias)
__global__ __launch_bounds__(128)
void k_gemm_relu_f16(const _Float16* __restrict__ A,
                     const _Float16* __restrict__ Bt,
                     const float* __restrict__ bias,
                     _Float16* __restrict__ C, int M, int N, int K) {
    int mtile  = blockIdx.y * 64 + (threadIdx.x >> 5) * 16;
    int ntile0 = blockIdx.x * 64;
    v8f acc[4];
    gemm_block(A, Bt, K, mtile, ntile0, acc);
    int n, moff; dcoords(n, moff);
#pragma unroll
    for (int j = 0; j < 4; ++j) {
        float bv = bias[ntile0 + j * 16 + n];
#pragma unroll
        for (int r = 0; r < 8; ++r)
            C[(size_t)(mtile + moff + r) * N + ntile0 + j * 16 + n] =
                (_Float16)fmaxf(acc[j][r] + bv, 0.f);
    }
}

// out = acc + bias + R32   (final output, f32)
__global__ __launch_bounds__(128)
void k_gemm_res_f32(const _Float16* __restrict__ A,
                    const _Float16* __restrict__ Bt,
                    const float* __restrict__ bias,
                    const float* __restrict__ R32,
                    float* __restrict__ C, int M, int N, int K) {
    int mtile  = blockIdx.y * 64 + (threadIdx.x >> 5) * 16;
    int ntile0 = blockIdx.x * 64;
    v8f acc[4];
    gemm_block(A, Bt, K, mtile, ntile0, acc);
    int n, moff; dcoords(n, moff);
#pragma unroll
    for (int j = 0; j < 4; ++j) {
        float bv = bias[ntile0 + j * 16 + n];
#pragma unroll
        for (int r = 0; r < 8; ++r) {
            size_t idx = (size_t)(mtile + moff + r) * N + ntile0 + j * 16 + n;
            C[idx] = acc[j][r] + bv + R32[idx];
        }
    }
}

// ---------------------------------------------------------------------------
// Flash attention: 1 wave per (b, h, 16-query tile). O = Q + softmax(QK^T/s)V
// ---------------------------------------------------------------------------
__global__ __launch_bounds__(32)
void k_attn(const _Float16* __restrict__ Q, const _Float16* __restrict__ Kf,
            const _Float16* __restrict__ Vt, float* __restrict__ O) {
    __shared__ __align__(16) _Float16 Pl[16 * 32];

    int qtile = blockIdx.x * 16;
    int h     = blockIdx.y;
    int b     = blockIdx.z;
    int n, moff; dcoords(n, moff);

    const _Float16* Qbase = Q + ((size_t)b * NQ + qtile) * DD + h * DH;
    const _Float16* Kbase = Kf + (size_t)b * NK * DD + h * DH;
    const _Float16* Vbase = Vt + ((size_t)b * HH + h) * DH * (size_t)NK;

    v16h qf = fragA_ld(Qbase, DD, 0, 0);

    float rmax[8], rsum[8];
    v8f o0 = {}, o1 = {};
#pragma unroll
    for (int r = 0; r < 8; ++r) { rmax[r] = -1e30f; rsum[r] = 0.f; }

    const float scale = 0.17677669529663687f;   // 1/sqrt(32)

    for (int kb = 0; kb < NK; kb += 32) {
        const _Float16* Kb = Kbase + (size_t)kb * DD;
        v16h kf0 = fragB_ld(Kb, DD, 0, 0);
        v16h kf1 = fragB_ld(Kb, DD, 16, 0);
        v8f z0 = {}, z1 = {};
        v8f s0 = wmma_f16(qf, kf0, z0);           // keys kb..kb+15
        v8f s1 = wmma_f16(qf, kf1, z1);           // keys kb+16..kb+31

#pragma unroll
        for (int r = 0; r < 8; ++r) {
            float a0 = s0[r] * scale, a1 = s1[r] * scale;
            float t  = fmaxf(a0, a1);
            t = fmaxf(t, __shfl_xor(t, 1, 32));
            t = fmaxf(t, __shfl_xor(t, 2, 32));
            t = fmaxf(t, __shfl_xor(t, 4, 32));
            t = fmaxf(t, __shfl_xor(t, 8, 32));
            float mnew = fmaxf(rmax[r], t);
            float corr = __expf(rmax[r] - mnew);
            float p0 = __expf(a0 - mnew);
            float p1 = __expf(a1 - mnew);
            float ps = p0 + p1;
            ps += __shfl_xor(ps, 1, 32);
            ps += __shfl_xor(ps, 2, 32);
            ps += __shfl_xor(ps, 4, 32);
            ps += __shfl_xor(ps, 8, 32);
            rsum[r] = rsum[r] * corr + ps;
            rmax[r] = mnew;
            o0[r] *= corr;
            o1[r] *= corr;
            int m = r + moff;
            Pl[m * 32 + n]      = (_Float16)p0;   // key col n
            Pl[m * 32 + 16 + n] = (_Float16)p1;   // key col 16+n
        }

        v16h pf  = fragA_ld(Pl, 32, 0, 0);        // P: 16 rows x 32 keys
        v16h vf0 = fragB_ld(Vbase, NK, 0, kb);    // dh 0..15
        v16h vf1 = fragB_ld(Vbase, NK, 16, kb);   // dh 16..31
        o0 = wmma_f16(pf, vf0, o0);
        o1 = wmma_f16(pf, vf1, o1);
    }

#pragma unroll
    for (int r = 0; r < 8; ++r) {
        int m = r + moff;
        size_t row = (size_t)b * NQ + qtile + m;
        float inv = 1.f / rsum[r];
        float q0 = (float)Q[row * DD + h * DH + n];
        float q1 = (float)Q[row * DD + h * DH + 16 + n];
        O[row * DD + h * DH + n]      = q0 + o0[r] * inv;
        O[row * DD + h * DH + 16 + n] = q1 + o1[r] * inv;
    }
}

// ---------------------------------------------------------------------------
// LayerNorm over D=256, one block (256 threads) per row.
// ---------------------------------------------------------------------------
__global__ void k_ln(const float* __restrict__ X, const float* __restrict__ g,
                     const float* __restrict__ bta, float* __restrict__ o32,
                     _Float16* __restrict__ o16) {
    __shared__ float red[256];
    int row = blockIdx.x, t = threadIdx.x;
    float x = X[(size_t)row * DD + t];
    red[t] = x;
    __syncthreads();
    for (int s = 128; s > 0; s >>= 1) {
        if (t < s) red[t] += red[t + s];
        __syncthreads();
    }
    float mean = red[0] * (1.f / DD);
    __syncthreads();
    float d = x - mean;
    red[t] = d * d;
    __syncthreads();
    for (int s = 128; s > 0; s >>= 1) {
        if (t < s) red[t] += red[t + s];
        __syncthreads();
    }
    float var = red[0] * (1.f / DD);
    float y = d * rsqrtf(var + 1e-5f) * g[t] + bta[t];
    if (o32) o32[(size_t)row * DD + t] = y;
    o16[(size_t)row * DD + t] = (_Float16)y;
}

// ---------------------------------------------------------------------------
// Launch
// ---------------------------------------------------------------------------
extern "C" void kernel_launch(void* const* d_in, const int* in_sizes, int n_in,
                              void* d_out, int out_size, void* d_ws, size_t ws_size,
                              hipStream_t stream) {
    const float* q  = (const float*)d_in[0];
    const float* x  = (const float*)d_in[1];
    const float* Wq = (const float*)d_in[2];  const float* bq = (const float*)d_in[3];
    const float* Wk = (const float*)d_in[4];  const float* bk = (const float*)d_in[5];
    const float* Wv = (const float*)d_in[6];  const float* bv = (const float*)d_in[7];
    const float* Wo = (const float*)d_in[8];  const float* bo = (const float*)d_in[9];
    const float* Wm = (const float*)d_in[10]; const float* bm = (const float*)d_in[11];
    const float* We = (const float*)d_in[12]; const float* be = (const float*)d_in[13];
    const float* g0 = (const float*)d_in[14]; const float* b0 = (const float*)d_in[15];
    const float* g1 = (const float*)d_in[16]; const float* b1 = (const float*)d_in[17];
    float* out = (float*)d_out;

    char* w = (char*)d_ws;
    size_t off = 0;
    auto take = [&](size_t bytes) -> void* {
        void* p = w + off;
        off = (off + bytes + 255) & ~(size_t)255;
        return p;
    };
    _Float16* qh   = (_Float16*)take((size_t)NTOK * DD * 2);
    _Float16* xh   = (_Float16*)take((size_t)NTOK * DD * 2);
    _Float16* WqT  = (_Float16*)take((size_t)DD * DD * 2);
    _Float16* WkT  = (_Float16*)take((size_t)DD * DD * 2);
    _Float16* WvT  = (_Float16*)take((size_t)DD * DD * 2);
    _Float16* WoT  = (_Float16*)take((size_t)DD * DD * 2);
    _Float16* WmT  = (_Float16*)take((size_t)DD * FF * 2);
    _Float16* WeT  = (_Float16*)take((size_t)FF * DD * 2);
    _Float16* Q16  = (_Float16*)take((size_t)NTOK * DD * 2);
    _Float16* K16  = (_Float16*)take((size_t)NTOK * DD * 2);
    _Float16* Vt16 = (_Float16*)take((size_t)NTOK * DD * 2);
    float*    Oatt = (float*)   take((size_t)NTOK * DD * 4);
    float*    ON32 = (float*)   take((size_t)NTOK * DD * 4);
    _Float16* ON16 = (_Float16*)take((size_t)NTOK * DD * 2);
    float*    X2   = (float*)   take((size_t)NTOK * DD * 4);
    _Float16* Y16  = (_Float16*)take((size_t)NTOK * DD * 2);
    _Float16* H16  = (_Float16*)take((size_t)NTOK * FF * 2);
    (void)ws_size; (void)in_sizes; (void)n_in; (void)out_size;

    const size_t nElemQX = (size_t)NTOK * DD;

    // 0) precision conversion + weight transposes
    k_cvt_f16<<<(nElemQX + 255) / 256, 256, 0, stream>>>(q, qh, nElemQX);
    k_cvt_f16<<<(nElemQX + 255) / 256, 256, 0, stream>>>(x, xh, nElemQX);
    dim3 tb(16, 16);
    k_tpose_f16<<<dim3(DD / 16, DD / 16), tb, 0, stream>>>(Wq, WqT, DD, DD);
    k_tpose_f16<<<dim3(DD / 16, DD / 16), tb, 0, stream>>>(Wk, WkT, DD, DD);
    k_tpose_f16<<<dim3(DD / 16, DD / 16), tb, 0, stream>>>(Wv, WvT, DD, DD);
    k_tpose_f16<<<dim3(DD / 16, DD / 16), tb, 0, stream>>>(Wo, WoT, DD, DD);
    k_tpose_f16<<<dim3(FF / 16, DD / 16), tb, 0, stream>>>(Wm, WmT, DD, FF);
    k_tpose_f16<<<dim3(DD / 16, FF / 16), tb, 0, stream>>>(We, WeT, FF, DD);

    // 1) Q/K/V projections (WMMA, 64x64 blocked, async-LDS staged B)
    dim3 gD(DD / 64, NTOK / 64);     // (4, 128)
    k_gemm_bias_f16<<<gD, 128, 0, stream>>>(qh, WqT, bq, Q16, NTOK, DD, DD);
    k_gemm_bias_f16<<<gD, 128, 0, stream>>>(xh, WkT, bk, K16, NTOK, DD, DD);
    k_gemm_vT     <<<gD, 128, 0, stream>>>(xh, WvT, bv, Vt16, NTOK, DD, DD);

    // 2) attention with online softmax + Q residual (WMMA)
    k_attn<<<dim3(NQ / 16, HH, BB), 32, 0, stream>>>(Q16, K16, Vt16, Oatt);

    // 3) LN(g0,b0)
    k_ln<<<NTOK, 256, 0, stream>>>(Oatt, g0, b0, ON32, ON16);

    // 4) X2 = ON + relu(ON @ Wo + bo)
    k_gemm_relu_res<<<gD, 128, 0, stream>>>(ON16, WoT, bo, ON32, X2, NTOK, DD, DD);

    // 5) LN(g1,b1) -> Y
    k_ln<<<NTOK, 256, 0, stream>>>(X2, g1, b1, nullptr, Y16);

    // 6) H = relu(Y @ Wm + bm)   (N=1024)
    dim3 gF(FF / 64, NTOK / 64);
    k_gemm_relu_f16<<<gF, 128, 0, stream>>>(Y16, WmT, bm, H16, NTOK, FF, DD);

    // 7) out = H @ We + be + X2   (K=1024)
    k_gemm_res_f32<<<gD, 128, 0, stream>>>(H16, WeT, be, X2, out, NTOK, DD, FF);
}
